// GCNModifiedAE2ASum_72756745994509
// MI455X (gfx1250) — matrix operand order repaired and checked by reading, not compiled, not run on hardware
//
#include <hip/hip_runtime.h>
#include <hip/hip_bf16.h>

// ---------------------------------------------------------------------------
// GCN forward on MI455X (gfx1250, wave32):
//   xw1 = x@w1 ; h1 = A@xw1 ; h1w2 = h1@w2 ; h2 = A@h1w2
//   a1 = h1@w1^T ; wc = w2^T@w1^T ; a2 = h2@wc
// Dense GEMMs: v_wmma_f32_16x16x32_bf16, A pre-converted to bf16 row-major,
// B pre-packed into WMMA B-fragment layout. Inner loop = b128 loads + WMMA.
// Sparse steps: f32 global atomics (h1/h2 stay resident in 192MB L2).
// ---------------------------------------------------------------------------

typedef __attribute__((ext_vector_type(16))) __bf16        v16bf;
typedef __attribute__((ext_vector_type(8)))  float         v8f;
typedef __attribute__((ext_vector_type(8)))  unsigned int  v8u;
typedef __attribute__((ext_vector_type(4)))  unsigned int  u4v;
typedef __attribute__((ext_vector_type(4)))  float         f4v;

__device__ __forceinline__ unsigned int pack_bf16x2(float lo, float hi) {
  unsigned int ul = __builtin_bit_cast(unsigned int, lo);
  unsigned int uh = __builtin_bit_cast(unsigned int, hi);
  ul = (ul + 0x7FFFu + ((ul >> 16) & 1u)) >> 16;   // round-to-nearest-even
  uh = (uh + 0x7FFFu + ((uh >> 16) & 1u)) >> 16;
  return ul | (uh << 16);
}

// ---------------------------------------------------------------------------
// f32 [n] -> packed bf16 dwords [n/2]; each thread converts 8 floats.
// All matrix sizes here are multiples of 8.
// ---------------------------------------------------------------------------
__global__ __launch_bounds__(256) void f32_to_bf16_kernel(
    const float* __restrict__ src, unsigned int* __restrict__ dst, long ndw) {
  long base = ((long)blockIdx.x * blockDim.x + threadIdx.x) * 4;
  if (base >= ndw) return;
  f4v a = *(const f4v*)(src + base * 2);
  f4v b = *(const f4v*)(src + base * 2 + 4);
  u4v o;
  o[0] = pack_bf16x2(a.x, a.y);
  o[1] = pack_bf16x2(a.z, a.w);
  o[2] = pack_bf16x2(b.x, b.y);
  o[3] = pack_bf16x2(b.z, b.w);
  *(u4v*)(dst + base) = o;
}

// ---------------------------------------------------------------------------
// Pack an f32 weight matrix (element (k,n) at src[k*sk + n*sn]) into the
// contiguous-K B-fragment layout used here for V_WMMA_F32_16X16X32_BF16:
//   fragment f = kt*(N/16)+nt ; per lane (N=lane%16, half=lane/16),
//   dword d holds bf16 pair for K = kt*32 + half*16 + 2d, 2d+1.
// ---------------------------------------------------------------------------
__global__ __launch_bounds__(256) void pack_b_kernel(
    const float* __restrict__ src, unsigned int* __restrict__ dst,
    int K, int N, int sk, int sn) {
  int i = blockIdx.x * blockDim.x + threadIdx.x;
  int frags = (K >> 5) * (N >> 4);
  if (i >= frags * 32) return;
  int lane = i & 31;
  int frag = i >> 5;
  int nfrag = N >> 4;
  int kt = frag / nfrag, nt = frag - kt * nfrag;
  int n = nt * 16 + (lane & 15);
  int half = lane >> 4;
  unsigned int* p = dst + (long)frag * 256 + lane * 8;
#pragma unroll
  for (int d = 0; d < 8; ++d) {
    int k = kt * 32 + half * 16 + d * 2;
    float lo = src[(long)k * sk + (long)n * sn];
    float hi = src[(long)(k + 1) * sk + (long)n * sn];
    p[d] = pack_bf16x2(lo, hi);
  }
}

// wc[i][j] = sum_m w2[m][i] * w1[j][m]   (i<128, j<512, m<256)  -- 33 MFLOP
__global__ void wc_kernel(const float* __restrict__ w1,
                          const float* __restrict__ w2,
                          float* __restrict__ wc) {
  int i = threadIdx.x;                              // 0..127 (H2)
  int j = blockIdx.x * blockDim.y + threadIdx.y;    // 0..511 (F_IN)
  float s = 0.f;
  for (int m = 0; m < 256; ++m)
    s = fmaf(w2[m * 128 + i], w1[(long)j * 256 + m], s);
  wc[(long)i * 512 + j] = s;
}

__global__ void zero_kernel(float* __restrict__ p, long n) {
  long i = (long)blockIdx.x * blockDim.x + threadIdx.x;
  if (i < n) p[i] = 0.f;
}

// ---------------------------------------------------------------------------
// C[M,N] = A_bf16[M,K] @ B_packed_bf16[K,N]
// One wave per 16x64 C strip: 4 WMMA accumulators share one A fragment.
// A is bf16 row-major (dword = 2 elems). Per K-step each lane loads two 16B
// chunks in contiguous K order matching the B packing (VGPR0-3: K=half*16+0..7,
// VGPR4-7: K=half*16+8..15). All branches wave-uniform -> EXEC all-ones.
// ---------------------------------------------------------------------------
__global__ __launch_bounds__(256) void gemm_bf16_wmma(
    const unsigned int* __restrict__ A, const unsigned int* __restrict__ Bpk,
    float* __restrict__ C, int M, int N, int K) {
  const int lane = threadIdx.x & 31;
  const int wave = threadIdx.x >> 5;
  const int tilesN4 = N >> 6;                        // quads of 16-col tiles
  int t = blockIdx.x * 8 + wave;                     // wave-uniform
  if (t >= (M >> 4) * tilesN4) return;
  const int mt = t / tilesN4;
  const int nq = t - mt * tilesN4;
  const int half = lane >> 4;
  const int m = lane & 15;
  const int nfrag = N >> 4;

  // A: dword offset = row*(K/2) + k0/2 + half*8 ; chunk1 at +4 dwords
  const unsigned int* arow = A + (long)(mt * 16 + m) * (K >> 1) + half * 8;
  const unsigned int* blane = Bpk + lane * 8 + (long)nq * 4 * 256;

  v8f acc0 = {}, acc1 = {}, acc2 = {}, acc3 = {};

#pragma unroll 2
  for (int k0 = 0; k0 < K; k0 += 32) {
    u4v alo = *(const u4v*)(arow + (k0 >> 1));
    u4v ahi = *(const u4v*)(arow + (k0 >> 1) + 4);
    __builtin_prefetch((const void*)(arow + (k0 >> 1) + 16), 0, 1);
    v8u ua = __builtin_shufflevector(alo, ahi, 0, 1, 2, 3, 4, 5, 6, 7);
    v16bf af = __builtin_bit_cast(v16bf, ua);

    const unsigned int* bp = blane + (long)(k0 >> 5) * nfrag * 256;
    v16bf bf0 = __builtin_bit_cast(v16bf, *(const v8u*)(bp));
    v16bf bf1 = __builtin_bit_cast(v16bf, *(const v8u*)(bp + 256));
    v16bf bf2 = __builtin_bit_cast(v16bf, *(const v8u*)(bp + 512));
    v16bf bf3 = __builtin_bit_cast(v16bf, *(const v8u*)(bp + 768));

    acc0 = __builtin_amdgcn_wmma_f32_16x16x32_bf16(
        false, af, false, bf0, (short)0, acc0, false, false);
    acc1 = __builtin_amdgcn_wmma_f32_16x16x32_bf16(
        false, af, false, bf1, (short)0, acc1, false, false);
    acc2 = __builtin_amdgcn_wmma_f32_16x16x32_bf16(
        false, af, false, bf2, (short)0, acc2, false, false);
    acc3 = __builtin_amdgcn_wmma_f32_16x16x32_bf16(
        false, af, false, bf3, (short)0, acc3, false, false);
  }

  // C/D layout: VGPR v -> row M = v + half*8, col N = lane%16
  float* c0 = C + (long)(mt * 16 + half * 8) * N + nq * 64 + m;
#pragma unroll
  for (int v = 0; v < 8; ++v) {
    c0[(long)v * N]      = acc0[v];
    c0[(long)v * N + 16] = acc1[v];
    c0[(long)v * N + 32] = acc2[v];
    c0[(long)v * N + 48] = acc3[v];
  }
}

// ---------------------------------------------------------------------------
// h[row[e], :] += val[e] * dense[col[e], :]   via global_atomic_add_f32
// One thread per (edge, feature); H is a power of two (hshift).
// ---------------------------------------------------------------------------
__global__ __launch_bounds__(256) void spmm_atomic_kernel(
    const int* __restrict__ rows, const int* __restrict__ cols,
    const float* __restrict__ vals, const float* __restrict__ dense,
    float* __restrict__ out, int hshift, int E) {
  long i = (long)blockIdx.x * blockDim.x + threadIdx.x;
  long total = (long)E << hshift;
  if (i >= total) return;
  int e = (int)(i >> hshift);
  int f = (int)(i & ((1 << hshift) - 1));
  int r = rows[e], c = cols[e];
  float v = vals[e];
  atomicAdd(out + ((long)r << hshift) + f, v * dense[((long)c << hshift) + f]);
}

// ---------------------------------------------------------------------------
extern "C" void kernel_launch(void* const* d_in, const int* in_sizes, int n_in,
                              void* d_out, int out_size, void* d_ws, size_t ws_size,
                              hipStream_t stream) {
  const float* x    = (const float*)d_in[0];   // [N, 512]
  const int*   arow = (const int*)d_in[1];     // [E]
  const int*   acol = (const int*)d_in[2];     // [E]
  const float* aval = (const float*)d_in[3];   // [E]
  const float* w1   = (const float*)d_in[4];   // [512, 256]
  const float* w2   = (const float*)d_in[5];   // [256, 128]

  const int FIN = 512, H1 = 256, H2 = 128;
  const int N = in_sizes[0] / FIN;             // 20000 (multiple of 16)
  const int E = in_sizes[1];                   // 640000

  // ---- workspace partition (bytes; 256B-aligned chunks) ----
  char* ws = (char*)d_ws;
  long off = 0;
  unsigned int* B1pk = (unsigned int*)(ws + off); off += (long)FIN * H1 * 2;   // 256KB
  unsigned int* B2pk = (unsigned int*)(ws + off); off += (long)H1 * H2 * 2;    //  64KB
  unsigned int* B3pk = (unsigned int*)(ws + off); off += (long)H1 * FIN * 2;   // 256KB
  unsigned int* B4pk = (unsigned int*)(ws + off); off += (long)H2 * FIN * 2;   // 128KB
  float*        wc   = (float*)       (ws + off); off += (long)H2 * FIN * 4;   // 256KB
  float*        xw1  = (float*)       (ws + off); off += (long)N * H1 * 4;
  float*        h1w2 = (float*)       (ws + off); off += (long)N * H2 * 4;
  unsigned int* xbf  = (unsigned int*)(ws + off); off += (long)N * FIN * 2;
  unsigned int* h1bf = (unsigned int*)(ws + off); off += (long)N * H1 * 2;
  unsigned int* h2bf = (unsigned int*)(ws + off); off += (long)N * H2 * 2;

  // ---- output partition (floats) ----
  float* a1 = (float*)d_out;                    // [N, 512]
  float* a2 = a1 + (long)N * FIN;               // [N, 512]
  float* h1 = a2 + (long)N * FIN;               // [N, 256]
  float* h2 = h1 + (long)N * H1;                // [N, 128]

  // 1) pack weights into WMMA B-fragment layout
  {
    int thr;
    thr = (FIN / 32) * (H1 / 16) * 32;
    pack_b_kernel<<<(thr + 255) / 256, 256, 0, stream>>>(w1, B1pk, FIN, H1, H1, 1);
    thr = (H1 / 32) * (H2 / 16) * 32;
    pack_b_kernel<<<(thr + 255) / 256, 256, 0, stream>>>(w2, B2pk, H1, H2, H2, 1);
    thr = (H1 / 32) * (FIN / 16) * 32;   // w1^T: element (k,n) = w1[n*256+k]
    pack_b_kernel<<<(thr + 255) / 256, 256, 0, stream>>>(w1, B3pk, H1, FIN, 1, H1);
  }

  // 2) wc = w2^T @ w1^T  (f32, tiny), then pack
  {
    dim3 blk(128, 2);
    wc_kernel<<<FIN / 2, blk, 0, stream>>>(w1, w2, wc);
    int thr = (H2 / 32) * (FIN / 16) * 32;
    pack_b_kernel<<<(thr + 255) / 256, 256, 0, stream>>>(wc, B4pk, H2, FIN, FIN, 1);
  }

  // 3) zero h1/h2 accumulation regions (d_out is poisoned)
  {
    long n = (long)N * (H1 + H2);
    zero_kernel<<<(int)((n + 255) / 256), 256, 0, stream>>>(h1, n);
  }

  // 4) xbf = bf16(x) ; xw1 = xbf @ w1
  {
    long ndw = (long)N * FIN / 2;
    f32_to_bf16_kernel<<<(int)((ndw / 4 + 255) / 256), 256, 0, stream>>>(x, xbf, ndw);
    int waves = (N / 16) * (H1 / 64);
    gemm_bf16_wmma<<<(waves + 7) / 8, 256, 0, stream>>>(xbf, B1pk, xw1, N, H1, FIN);
  }

  // 5) h1 = A @ xw1          (scatter-add, H=256 -> hshift 8)
  spmm_atomic_kernel<<<(int)(((long)E << 8) / 256), 256, 0, stream>>>(
      arow, acol, aval, xw1, h1, 8, E);

  // 6) h1bf = bf16(h1) ; h1w2 = h1bf @ w2
  {
    long ndw = (long)N * H1 / 2;
    f32_to_bf16_kernel<<<(int)((ndw / 4 + 255) / 256), 256, 0, stream>>>(h1, h1bf, ndw);
    int waves = (N / 16) * (H2 / 64);
    gemm_bf16_wmma<<<(waves + 7) / 8, 256, 0, stream>>>(h1bf, B2pk, h1w2, N, H2, H1);
  }

  // 7) h2 = A @ h1w2         (H=128 -> hshift 7)
  spmm_atomic_kernel<<<(int)(((long)E << 7) / 256), 256, 0, stream>>>(
      arow, acol, aval, h1w2, h2, 7, E);

  // 8) h2bf = bf16(h2)
  {
    long ndw = (long)N * H2 / 2;
    f32_to_bf16_kernel<<<(int)((ndw / 4 + 255) / 256), 256, 0, stream>>>(h2, h2bf, ndw);
  }

  // 9) a1 = h1bf @ w1^T      [20000,256]x[256,512]
  {
    int waves = (N / 16) * (FIN / 64);
    gemm_bf16_wmma<<<(waves + 7) / 8, 256, 0, stream>>>(h1bf, B3pk, a1, N, FIN, H1);
  }

  // 10) a2 = h2bf @ wc       [20000,128]x[128,512]
  {
    int waves = (N / 16) * (FIN / 64);
    gemm_bf16_wmma<<<(waves + 7) / 8, 256, 0, stream>>>(h2bf, B4pk, a2, N, FIN, H2);
  }
}
